// BaseSparseVAE_13194139533578
// MI455X (gfx1250) — compile-verified, matrix operand order checked by reading, not run on hardware
//
#include <hip/hip_runtime.h>
#include <hip/hip_bf16.h>
#include <math.h>

typedef __attribute__((ext_vector_type(16))) _Float16 v16h;
typedef __attribute__((ext_vector_type(8)))  _Float16 v8h;
typedef __attribute__((ext_vector_type(8)))  float    v8f;
typedef __attribute__((ext_vector_type(4)))  float    v4f;

#define B_   256
#define D_   512
#define L_   64
#define H_   300
#define HPAD 304   // H padded to multiple of 16 (N-tiles: 19)
#define KPAD 320   // H padded to multiple of 32 (K-steps: 10)

// LDS row strides (halves / floats). f16 tile strides chosen so that
// stride_dwords mod 64 == 4*odd -> the 16 lo-lanes of a b128 fragment load
// land on 16 disjoint 4-bank groups (minimal 2-phase LDS access).
#define SM_LD  72    // sMask  (36 dwords = 4*9)
#define SW1_LD 72    // sW1
#define SG1_LD 328   // sG1    (164 dwords -> mod 64 = 36 = 4*9)
#define SW2_LD 328   // sW2
#define SHW_LD 304   // sHw (f32)

// ---------------------------------------------------------------------------
// Fragment loader for 16-bit WMMA A/B operands (CDNA5 ISA 7.12.2 layout):
//   lane 0-15  : M (or N) = lane, holds K = 0..7 (v0-3) and 16..23 (v4-7)
//   lane 16-31 : M (or N) = lane-16, holds K = 8..15 and 24..31
// Two ds_load_b128 per fragment. B operands use the same loader on the
// [N,K] row-major (i.e. B^T) weight tile.
// ---------------------------------------------------------------------------
__device__ __forceinline__ v16h load_frag16(const _Float16* base, int ld, int lane) {
  const int m  = lane & 15;
  const int hi = (lane >> 4) & 1;
  const _Float16* p = base + m * ld + hi * 8;
  v8h lo = *(const v8h*)(p);        // K = hi*8 + 0..7
  v8h hp = *(const v8h*)(p + 16);   // K = 16 + hi*8 + 0..7
  v16h r;
#pragma unroll
  for (int i = 0; i < 8; ++i) { r[i] = lo[i]; r[i + 8] = hp[i]; }
  return r;
}

// ---------------------------------------------------------------------------
// One-shot weight pre-pack: f32 -> f16, zero-padded to [HPAD][64] / [HPAD][KPAD]
// so 2048 decoder blocks do pure b128 copies instead of repeated cvt work.
// ---------------------------------------------------------------------------
__global__ void pack_weights_kernel(const float* __restrict__ gen1_w,
                                    const float* __restrict__ gen2_w,
                                    _Float16* __restrict__ wsW1,   // [HPAD][64]
                                    _Float16* __restrict__ wsW2) { // [HPAD][KPAD]
  int idx = blockIdx.x * blockDim.x + threadIdx.x;
  const int N1 = HPAD * 64;
  const int N2 = HPAD * KPAD;
  if (idx < N1) {
    int r = idx >> 6, c = idx & 63;
    wsW1[idx] = (r < H_) ? (_Float16)gen1_w[r * L_ + c] : (_Float16)0.f;
  } else if (idx < N1 + N2) {
    int i2 = idx - N1;
    int r = i2 / KPAD, c = i2 % KPAD;
    wsW2[i2] = (r < H_ && c < H_) ? (_Float16)gen2_w[r * H_ + c] : (_Float16)0.f;
  }
}

// ---------------------------------------------------------------------------
// Fused decoder: one workgroup = 4 b-values x 16 d-values = 64 GEMM rows.
//   masked = z[b]*W[d]            [64 x 64]   f16 LDS
//   g1 = relu(masked @ gen1_w^T)  [64 x 300]  f16 LDS (K=64 -> 2 WMMA)
//   g2 = relu(g1 @ gen2_w^T + b)  [64 x 300]  (K=320 -> 10 WMMA)
//   x_mean[b,d] = dot(g2, head_w[d]) + head_b[d]  (fused shfl-reduce)
// Waves process tile PAIRS -> two independent WMMA accumulator chains to
// cover the 4-coexec f16-WMMA RAW hazard slots.
// ---------------------------------------------------------------------------
__global__ __launch_bounds__(256)
void decoder_fused(const float* __restrict__ z,
                   const float* __restrict__ Wmask,
                   const _Float16* __restrict__ wsW1,
                   const _Float16* __restrict__ wsW2,
                   const float* __restrict__ gen2_b,
                   const float* __restrict__ head_w,   // [D, H]
                   const float* __restrict__ head_b,
                   float* __restrict__ x_mean) {
  __shared__ __align__(16) _Float16 sMask[64][SM_LD];     //   9.0 KB
  __shared__ __align__(16) _Float16 sG1[64][SG1_LD];      //  41.0 KB
  __shared__ __align__(16) _Float16 sW1[HPAD][SW1_LD];    //  42.8 KB
  __shared__ __align__(16) _Float16 sW2[HPAD][SW2_LD];    // 194.8 KB
  __shared__ __align__(16) float    sHw[16][SHW_LD];      //  19.0 KB (head_w tile)
  __shared__ float sGb[HPAD];                             //   1.2 KB (zero-padded bias)
  __shared__ float sXm[8][64];                            //   2.0 KB

  const int tid    = threadIdx.x;
  const int lane   = tid & 31;
  const int wave   = tid >> 5;
  const int b_base = blockIdx.y * 4;
  const int d_base = blockIdx.x * 16;

  // ---- preamble: stage everything in LDS ----------------------------------
  for (int idx = tid; idx < 64 * 64; idx += 256) {              // masked (scalar)
    int m = idx >> 6, l = idx & 63;                             // m = bi*16 + di
    float v = z[(b_base + (m >> 4)) * L_ + l] * Wmask[(d_base + (m & 15)) * L_ + l];
    sMask[m][l] = (_Float16)v;
  }
  for (int idx = tid; idx < HPAD * 8; idx += 256) {             // gen1_w, b128 copies
    int r = idx >> 3, g = idx & 7;
    *(v8h*)(&sW1[r][g * 8]) = *(const v8h*)(wsW1 + r * 64 + g * 8);
  }
  for (int idx = tid; idx < HPAD * (KPAD / 8); idx += 256) {    // gen2_w, b128 copies
    int r = idx / (KPAD / 8), g = idx % (KPAD / 8);
    *(v8h*)(&sW2[r][g * 8]) = *(const v8h*)(wsW2 + r * KPAD + g * 8);
  }
  for (int idx = tid; idx < 16 * (HPAD / 4); idx += 256) {      // head_w tile, float4
    int r = idx / (HPAD / 4), c4 = idx % (HPAD / 4);
    if (c4 < H_ / 4) {
      *(v4f*)(&sHw[r][c4 * 4]) = *(const v4f*)(head_w + (d_base + r) * H_ + c4 * 4);
    } else {
      v4f zz = {};
      *(v4f*)(&sHw[r][c4 * 4]) = zz;                            // cols 300..303 = 0
    }
  }
  for (int idx = tid; idx < HPAD; idx += 256)                   // zero-padded gen2_b
    sGb[idx] = (idx < H_) ? gen2_b[idx] : 0.f;
  for (int idx = tid; idx < 64 * 16; idx += 256)                // zero K-pad of g1
    sG1[idx >> 4][HPAD + (idx & 15)] = (_Float16)0.f;
  for (int idx = tid; idx < 8 * 64; idx += 256) sXm[idx >> 6][idx & 63] = 0.f;
  __syncthreads();

  // ---- stage 1: g1 = relu(masked @ gen1_w^T), K=64, dual-tile chains ------
  for (int t0 = 2 * wave; t0 < 4 * 19; t0 += 16) {
    const int t1 = t0 + 1;
    const int bi0 = t0 / 19, nt0 = t0 % 19;
    const int bi1 = t1 / 19, nt1 = t1 % 19;
    v16h a00 = load_frag16(&sMask[bi0 * 16][0],  SM_LD, lane);
    v16h a01 = load_frag16(&sMask[bi0 * 16][32], SM_LD, lane);
    v16h b00 = load_frag16(&sW1[nt0 * 16][0],    SW1_LD, lane);
    v16h b01 = load_frag16(&sW1[nt0 * 16][32],   SW1_LD, lane);
    v16h a10 = load_frag16(&sMask[bi1 * 16][0],  SM_LD, lane);
    v16h a11 = load_frag16(&sMask[bi1 * 16][32], SM_LD, lane);
    v16h b10 = load_frag16(&sW1[nt1 * 16][0],    SW1_LD, lane);
    v16h b11 = load_frag16(&sW1[nt1 * 16][32],   SW1_LD, lane);
    v8f acc0 = {}, acc1 = {};
    acc0 = __builtin_amdgcn_wmma_f32_16x16x32_f16(false, a00, false, b00, (short)0, acc0, false, false);
    acc1 = __builtin_amdgcn_wmma_f32_16x16x32_f16(false, a10, false, b10, (short)0, acc1, false, false);
    acc0 = __builtin_amdgcn_wmma_f32_16x16x32_f16(false, a01, false, b01, (short)0, acc0, false, false);
    acc1 = __builtin_amdgcn_wmma_f32_16x16x32_f16(false, a11, false, b11, (short)0, acc1, false, false);
    const int n = lane & 15, hi = lane >> 4;
#pragma unroll
    for (int r = 0; r < 8; ++r) {
      sG1[bi0 * 16 + r + hi * 8][nt0 * 16 + n] = (_Float16)fmaxf(acc0[r], 0.f);
      sG1[bi1 * 16 + r + hi * 8][nt1 * 16 + n] = (_Float16)fmaxf(acc1[r], 0.f);
    }
  }
  __syncthreads();

  // ---- stage 2 + fused head, dual-tile chains -----------------------------
  auto tail = [&](const v8f& acc, int bi, int nt) {
    const int n = lane & 15, hi = lane >> 4;
    const int h = nt * 16 + n;                   // >=300 lanes: sGb/sHw pads give 0
    const float bias = sGb[h];
#pragma unroll
    for (int r = 0; r < 8; ++r) {
      const int di = r + hi * 8;                 // m_local = bi*16 + di
      float g2 = fmaxf(acc[r] + bias, 0.f);
      float v  = g2 * sHw[di][h];
      v += __shfl_xor(v, 1, 32);
      v += __shfl_xor(v, 2, 32);
      v += __shfl_xor(v, 4, 32);
      v += __shfl_xor(v, 8, 32);
      if ((lane & 15) == 0) sXm[wave][bi * 16 + di] += v;   // single writer/slot
    }
  };

  for (int t0 = 2 * wave; t0 < 4 * 19; t0 += 16) {
    const int t1 = t0 + 1;
    const int bi0 = t0 / 19, nt0 = t0 % 19;
    const int bi1 = t1 / 19, nt1 = t1 % 19;
    v8f acc0 = {}, acc1 = {};
#pragma unroll 2
    for (int kt = 0; kt < KPAD / 32; ++kt) {
      v16h a0 = load_frag16(&sG1[bi0 * 16][kt * 32], SG1_LD, lane);
      v16h b0 = load_frag16(&sW2[nt0 * 16][kt * 32], SW2_LD, lane);
      acc0 = __builtin_amdgcn_wmma_f32_16x16x32_f16(false, a0, false, b0, (short)0, acc0, false, false);
      v16h a1 = load_frag16(&sG1[bi1 * 16][kt * 32], SG1_LD, lane);
      v16h b1 = load_frag16(&sW2[nt1 * 16][kt * 32], SW2_LD, lane);
      acc1 = __builtin_amdgcn_wmma_f32_16x16x32_f16(false, a1, false, b1, (short)0, acc1, false, false);
    }
    tail(acc0, bi0, nt0);
    tail(acc1, bi1, nt1);
  }
  __syncthreads();

  // ---- epilogue: deterministic cross-wave reduction + bias ----------------
  if (tid < 64) {
    float s = 0.f;
#pragma unroll
    for (int w = 0; w < 8; ++w) s += sXm[w][tid];
    int b = b_base + (tid >> 4);
    int d = d_base + (tid & 15);
    x_mean[b * D_ + d] = s + head_b[d];
  }
}

// ---------------------------------------------------------------------------
// Tiny encoder kernels (0.13 GFLOP total — <1% of work; plain fp32 is fine)
// ---------------------------------------------------------------------------
__global__ void dense_relu_kernel(const float* __restrict__ in,
                                  const float* __restrict__ w,   // [N,K]
                                  const float* __restrict__ bias,
                                  float* __restrict__ out,
                                  int M, int N, int K) {
  int idx = blockIdx.x * blockDim.x + threadIdx.x;
  if (idx >= M * N) return;
  int m = idx / N, n = idx % N;
  const float* ip = in + m * K;
  const float* wp = w + n * K;
  float s = bias[n];
  for (int k = 0; k < K; ++k) s += ip[k] * wp[k];
  out[idx] = fmaxf(s, 0.f);
}

__global__ void latent_kernel(const float* __restrict__ h2,
                              const float* __restrict__ eps,
                              const float* __restrict__ zm_w, const float* __restrict__ zm_b,
                              const float* __restrict__ zv_w, const float* __restrict__ zv_b,
                              float* __restrict__ z_mean, float* __restrict__ z_logvar,
                              float* __restrict__ z_out,  float* __restrict__ z_ws) {
  int idx = blockIdx.x * blockDim.x + threadIdx.x;
  if (idx >= B_ * L_) return;
  int b = idx >> 6, l = idx & 63;
  const float* hp = h2 + b * H_;
  const float* mp = zm_w + l * H_;
  const float* vp = zv_w + l * H_;
  float sm = zm_b[l], sv = zv_b[l];
  for (int k = 0; k < H_; ++k) { float h = hp[k]; sm += h * mp[k]; sv += h * vp[k]; }
  float zz = sm + eps[idx] * expf(0.5f * sv);
  z_mean[idx] = sm; z_logvar[idx] = sv; z_out[idx] = zz; z_ws[idx] = zz;
}

// ---------------------------------------------------------------------------
extern "C" void kernel_launch(void* const* d_in, const int* in_sizes, int n_in,
                              void* d_out, int out_size, void* d_ws, size_t ws_size,
                              hipStream_t stream) {
  const float* x      = (const float*)d_in[0];
  const float* eps    = (const float*)d_in[1];
  const float* Wmask  = (const float*)d_in[2];
  const float* enc1_w = (const float*)d_in[3];
  const float* enc1_b = (const float*)d_in[4];
  const float* enc2_w = (const float*)d_in[5];
  const float* enc2_b = (const float*)d_in[6];
  const float* zm_w   = (const float*)d_in[7];
  const float* zm_b   = (const float*)d_in[8];
  const float* zv_w   = (const float*)d_in[9];
  const float* zv_b   = (const float*)d_in[10];
  const float* gen1_w = (const float*)d_in[11];
  const float* gen2_w = (const float*)d_in[12];
  const float* gen2_b = (const float*)d_in[13];
  const float* head_w = (const float*)d_in[14];
  const float* head_b = (const float*)d_in[15];

  float* out      = (float*)d_out;
  float* x_mean_o = out;                         // [256*512]
  float* z_o      = out + B_ * D_;               // [256*64]
  float* zm_o     = z_o + B_ * L_;               // [256*64]
  float* zlv_o    = zm_o + B_ * L_;              // [256*64]

  float* ws   = (float*)d_ws;
  float* h1   = ws;                              // [256*300]  f32
  float* h2   = h1 + B_ * H_;                    // [256*300]  f32
  float* zws  = h2 + B_ * H_;                    // [256*64]   f32
  _Float16* wsW1 = (_Float16*)(zws + B_ * L_);   // [HPAD*64]  f16 (16B-aligned)
  _Float16* wsW2 = wsW1 + HPAD * 64;             // [HPAD*KPAD] f16

  // weight pre-pack (f32 -> padded f16 images), once per launch
  {
    int n = HPAD * 64 + HPAD * KPAD;
    pack_weights_kernel<<<(n + 255) / 256, 256, 0, stream>>>(gen1_w, gen2_w, wsW1, wsW2);
  }
  // encoder
  {
    int n = B_ * H_;
    dense_relu_kernel<<<(n + 255) / 256, 256, 0, stream>>>(x, enc1_w, enc1_b, h1,
                                                           B_, H_, D_);
    dense_relu_kernel<<<(n + 255) / 256, 256, 0, stream>>>(h1, enc2_w, enc2_b, h2,
                                                           B_, H_, H_);
  }
  // latents
  {
    int n = B_ * L_;
    latent_kernel<<<(n + 255) / 256, 256, 0, stream>>>(h2, eps, zm_w, zm_b, zv_w, zv_b,
                                                       zm_o, zlv_o, z_o, zws);
  }
  // fused WMMA decoder: grid = (D/16, B/4) = (32, 64)
  {
    dim3 grid(D_ / 16, B_ / 4);
    decoder_fused<<<grid, 256, 0, stream>>>(zws, Wmask, wsW1, wsW2, gen2_b,
                                            head_w, head_b, x_mean_o);
  }
}
